// Mamba2Layer_10617159156220
// MI455X (gfx1250) — compile-verified, matrix operand order checked by reading, not run on hardware
//
#include <hip/hip_runtime.h>

#define L_SEQ    2048
#define D_MODEL  1024
#define D_INNER  2048
#define N_HEADS  16
#define D_STATE  64
#define D_HEAD   128
#define CONV_DIM 2176          // D_INNER + 2*D_STATE
#define D_WIN    4240          // 2*D_INNER + 2*D_STATE + N_HEADS
#define DT_COL   4224          // D_INNER + CONV_DIM

typedef __attribute__((ext_vector_type(16))) __bf16 v16bf;
typedef __attribute__((ext_vector_type(8)))  float  v8f;
typedef __attribute__((ext_vector_type(4)))  unsigned int u32x4;
typedef __attribute__((ext_vector_type(8)))  int  i32x8;
typedef __attribute__((ext_vector_type(4)))  int  i32x4;

// ---------- helpers ----------------------------------------------------------

__device__ __forceinline__ __bf16 to_bf16(float f) {
  union { float f; unsigned u; } a; a.f = f;
  unsigned r = a.u + 0x7FFFu + ((a.u >> 16) & 1u);   // round-to-nearest-even
  union { unsigned short s; __bf16 h; } b;
  b.s = (unsigned short)(r >> 16);
  return b.h;
}

// A-fragment (16x32, 16-bit) per ISA 7.12.2: lane m = lane&15 holds row m,
// K-chunks {hi*8 .. hi*8+7} and {16+hi*8 .. 16+hi*8+7}  (hi = lane>>4).
// B-fragment is the same layout with N playing the role of M, so when the
// B matrix is stored N-major (Bt row n = column n), the SAME loader works.
__device__ __forceinline__ v16bf load_frag(const __bf16* base, int ld, int row0, int k0) {
  int lane = threadIdx.x & 31;
  int m = lane & 15, hi = lane >> 4;
  const __bf16* p = base + (size_t)(row0 + m) * ld + k0 + hi * 8;
  union { uint4 q[2]; v16bf v; } u;
  u.q[0] = *(const uint4*)p;          // K = k0 + hi*8 .. +7
  u.q[1] = *(const uint4*)(p + 16);   // K = k0 + 16 + hi*8 .. +7
  return u.v;
}

__device__ __forceinline__ v8f wmma_bf16(v16bf a, v16bf b, v8f c) {
  // (neg_a, A, neg_b, B, c_mod, C, reuse_a, reuse_b)
  return __builtin_amdgcn_wmma_f32_16x16x32_bf16(false, a, false, b, (short)0, c,
                                                 false, false);
}

// TDM: DMA a 2D tile (tile_w x tile_h elements, 2-byte data) from a row-major
// tensor in global memory into LDS.  D# packing per ISA 8.3/8.4:
//   group0: count=1 | lds_addr | global_addr[56:0] | type=2
//   group1: data_size=1(2B), tensor_dim0/1, tile_dim0/1 (tile_dim2=0 -> 2D),
//           tensor_dim0_stride
//   groups 2/3 (+ extra group on clang-23's 6-arg builtin): zero (2D tiles)
__device__ __forceinline__ void tdm_load_tile_2d(const void* gaddr, unsigned lds_off,
                                                 int width, int height, int stride,
                                                 int tile_w, int tile_h) {
  unsigned long long ga = (unsigned long long)(uintptr_t)gaddr;
  u32x4 g0;
  g0[0] = 1u;                                                  // count=1, user D#
  g0[1] = lds_off;                                             // lds_addr (bytes)
  g0[2] = (unsigned)ga;                                        // global_addr[31:0]
  g0[3] = ((unsigned)(ga >> 32) & 0x01FFFFFFu) | 0x80000000u;  // addr[56:32]|type=2
  i32x8 g1;
  g1[0] = 0x00010000;                            // workgroup_mask=0, data_size=1 (2B)
  g1[1] = (width & 0xFFFF) << 16;                // tensor_dim0[15:0]  @ bits 63:48
  g1[2] = ((unsigned)width >> 16) | ((height & 0xFFFF) << 16);  // dim0 hi | dim1 lo
  g1[3] = ((unsigned)height >> 16) | (tile_w << 16);            // dim1 hi | tile_dim0
  g1[4] = tile_h & 0xFFFF;                       // tile_dim1 (tile_dim2 = 0 -> 2D)
  g1[5] = stride;                                // tensor_dim0_stride[31:0]
  g1[6] = 0;
  g1[7] = 0;
  i32x4 gz4; gz4[0] = 0; gz4[1] = 0; gz4[2] = 0; gz4[3] = 0;
  i32x8 gz8;
#pragma unroll
  for (int i = 0; i < 8; ++i) gz8[i] = 0;
  // clang-23 (therock-10.0) 6-arg form: (g0, g1, g2, g3, g4, cpol)
  __builtin_amdgcn_tensor_load_to_lds(g0, g1, gz4, gz4, gz8, 0);
}

// ---------- kernels ----------------------------------------------------------

__global__ void k_f32_to_bf16(const float* in, __bf16* out, int n) {
  int i = blockIdx.x * 256 + threadIdx.x;
  if (i < n) out[i] = to_bf16(in[i]);
}

__global__ void k_rmsnorm_in(const float* x, const float* w, __bf16* out) {
  int t = blockIdx.x, tid = threadIdx.x;
  const float* row = x + (size_t)t * D_MODEL;
  float v[4]; float ss = 0.f;
#pragma unroll
  for (int i = 0; i < 4; ++i) { v[i] = row[tid + i * 256]; ss += v[i] * v[i]; }
  __shared__ float red[256];
  red[tid] = ss; __syncthreads();
  for (int off = 128; off > 0; off >>= 1) {
    if (tid < off) red[tid] += red[tid + off];
    __syncthreads();
  }
  float inv = rsqrtf(red[0] / (float)D_MODEL + 1e-5f);
#pragma unroll
  for (int i = 0; i < 4; ++i) {
    int c = tid + i * 256;
    out[(size_t)t * D_MODEL + c] = to_bf16(v[i] * inv * w[c]);
  }
}

// Statically specialized GEMM body: NV n-tiles per wave, fully unrolled so the
// accumulators live in fixed VGPRs (no v_movrel indexed-register traffic).
// K must be a multiple of 64.
template <int NV>
__device__ __forceinline__ void gemm_body(const __bf16* __restrict__ A,
                                          const __bf16* __restrict__ Bt,
                                          float* __restrict__ C,
                                          int N, int K, int row0, int nt0) {
  int lane = threadIdx.x & 31;
  int m = lane & 15, hi = lane >> 4;

  v8f acc[NV];
#pragma unroll
  for (int j = 0; j < NV; ++j)
#pragma unroll
    for (int e = 0; e < 8; ++e) acc[j][e] = 0.f;

  for (int k = 0; k < K; k += 64) {   // k-unroll by 2
    v16bf a0 = load_frag(A, K, row0, k);
    v16bf a1 = load_frag(A, K, row0, k + 32);
    v16bf b0[NV], b1[NV];
#pragma unroll
    for (int j = 0; j < NV; ++j) b0[j] = load_frag(Bt, K, (nt0 + j) * 16, k);
#pragma unroll
    for (int j = 0; j < NV; ++j) b1[j] = load_frag(Bt, K, (nt0 + j) * 16, k + 32);
#pragma unroll
    for (int j = 0; j < NV; ++j) acc[j] = wmma_bf16(a0, b0[j], acc[j]);
#pragma unroll
    for (int j = 0; j < NV; ++j) acc[j] = wmma_bf16(a1, b1[j], acc[j]);
  }

#pragma unroll
  for (int j = 0; j < NV; ++j)
#pragma unroll
    for (int e = 0; e < 8; ++e) {
      int r = row0 + e + 8 * hi;           // D layout: VGPR e -> row e + 8*hi
      int c = (nt0 + j) * 16 + m;          // col = lane&15
      C[(size_t)r * N + c] = acc[j][e];
    }
}

// C[M,N] = A[M,K] * Bt[N,K]^T   (Bt is N-major: row n == column n of B)
__global__ void k_gemm_bf16(const __bf16* __restrict__ A,
                            const __bf16* __restrict__ Bt,
                            float* __restrict__ C, int M, int N, int K) {
  int wave = threadIdx.x >> 5;
  int mtile = blockIdx.x * 4 + wave;
  if (mtile * 16 >= M) return;
  int row0 = mtile * 16;
  int ntiles = N >> 4;
  int nt0 = blockIdx.y * 4;
  int nvalid = ntiles - nt0; if (nvalid > 4) nvalid = 4;

  if (nvalid == 4) {
    gemm_body<4>(A, Bt, C, N, K, row0, nt0);
  } else {
    for (int j = 0; j < nvalid; ++j)      // remainder: straight-line NV=1 body
      gemm_body<1>(A, Bt, C, N, K, row0, nt0 + j);
  }
}

__global__ void k_conv_silu(const float* zx, const float* conv, const float* cbias,
                            float* xBCc) {
  int i = blockIdx.x * 256 + threadIdx.x;
  if (i >= L_SEQ * CONV_DIM) return;
  int t = i / CONV_DIM, c = i - t * CONV_DIM;
  float acc = cbias[c];
#pragma unroll
  for (int mm = 0; mm < 4; ++mm) {
    int tm = t - mm;
    if (tm >= 0) acc += zx[(size_t)tm * D_WIN + D_INNER + c] * conv[c * 4 + mm];
  }
  xBCc[i] = acc / (1.f + __expf(-acc));   // SiLU
}

__global__ void k_dt_cum(const float* zx, const float* dt_bias, const float* A_log,
                         float* dt_sp, float* cumA) {
  int h = threadIdx.x;
  if (h >= N_HEADS) return;
  float bias = dt_bias[h];
  float negA = -__expf(A_log[h]);
  float run = 0.f;
  for (int t = 0; t < L_SEQ; ++t) {
    float v = zx[(size_t)t * D_WIN + DT_COL + h] + bias;
    float sp = (v > 20.f) ? v : log1pf(__expf(v));     // softplus
    dt_sp[t * N_HEADS + h] = sp;
    run += negA * sp;                                   // inclusive cumsum of Adt
    cumA[h * L_SEQ + t] = run;
  }
}

__global__ void k_prep(const float* xBCc, const float* dt_sp,
                       __bf16* xdtT, __bf16* Bb, __bf16* Cb) {
  int i = blockIdx.x * 256 + threadIdx.x;
  if (i >= L_SEQ * CONV_DIM) return;
  int t = i / CONV_DIM, c = i - t * CONV_DIM;
  float v = xBCc[i];
  if (c < D_INNER) {
    float s = dt_sp[t * N_HEADS + (c >> 7)];
    xdtT[(size_t)c * L_SEQ + t] = to_bf16(v * s);       // transposed for B-fragments
  } else if (c < D_INNER + D_STATE) {
    Bb[t * D_STATE + (c - D_INNER)] = to_bf16(v);
  } else {
    Cb[t * D_STATE + (c - D_INNER - D_STATE)] = to_bf16(v);
  }
}

// One wave per (head, 16-row t-tile). Chunked quadratic form:
//   G(16x32) = C_tile(16x64) * B_rows^T  -> decay/mask -> bf16 via LDS ->
//   Y(16x128) += M(16x32) * xdt(32x128)
// The xdt 128x32 B-operand tile is DMAed into LDS by the Tensor Data Mover
// while the wave computes G and the decay/mask, then consumed as B-fragments.
__global__ void k_ssm(const __bf16* Cb, const __bf16* Bb, const __bf16* xdtT,
                      const float* cumA, const float* Dvec, const float* xBCc,
                      float* y) {
  __shared__ __attribute__((aligned(16))) __bf16 Mtile[16 * 32];      // 1 KB
  __shared__ __attribute__((aligned(16))) __bf16 Xtile[D_HEAD * 32];  // 8 KB
  int tb = blockIdx.x, h = blockIdx.y;
  int lane = threadIdx.x & 31;
  int m = lane & 15, hi = lane >> 4;
  int tbase = tb * 16;
  const float* cA = cumA + h * L_SEQ;
  unsigned xtile_lds = (unsigned)(uintptr_t)&Xtile[0];   // LDS offset = addr[31:0]

  v16bf cf0 = load_frag(Cb, D_STATE, tbase, 0);    // C cols 0..31
  v16bf cf1 = load_frag(Cb, D_STATE, tbase, 32);   // C cols 32..63
  float cat[8];
#pragma unroll
  for (int e = 0; e < 8; ++e) cat[e] = cA[tbase + e + 8 * hi];

  v8f acc[8];
#pragma unroll
  for (int j = 0; j < 8; ++j)
#pragma unroll
    for (int e = 0; e < 8; ++e) acc[j][e] = 0.f;
  v8f zero;
#pragma unroll
  for (int e = 0; e < 8; ++e) zero[e] = 0.f;

  for (int sb = 0; sb <= tb; sb += 2) {            // two 16-wide s-subtiles / iter
    int s0 = sb * 16;

    // WAR: previous iteration's LDS fragment reads must drain before the TDM
    // overwrites Xtile; then kick off the async DMA of xdtT[h*128..+128, s0..+32].
    asm volatile("s_wait_dscnt 0" ::: "memory");
    tdm_load_tile_2d(xdtT + (size_t)h * D_HEAD * L_SEQ + s0, xtile_lds,
                     L_SEQ - s0, D_HEAD, L_SEQ, 32, D_HEAD);

    v8f g0 = wmma_bf16(cf0, load_frag(Bb, D_STATE, s0, 0), zero);
    g0     = wmma_bf16(cf1, load_frag(Bb, D_STATE, s0, 32), g0);
    float cs0 = cA[s0 + m];
#pragma unroll
    for (int e = 0; e < 8; ++e) {
      int t = tbase + e + 8 * hi;
      int s = s0 + m;
      float val = (s <= t) ? __expf(cat[e] - cs0) * g0[e] : 0.f;
      Mtile[(e + 8 * hi) * 32 + m] = to_bf16(val);
    }
    if (sb + 1 <= tb) {
      int s1 = s0 + 16;
      v8f g1 = wmma_bf16(cf0, load_frag(Bb, D_STATE, s1, 0), zero);
      g1     = wmma_bf16(cf1, load_frag(Bb, D_STATE, s1, 32), g1);
      float cs1 = cA[s1 + m];
#pragma unroll
      for (int e = 0; e < 8; ++e) {
        int t = tbase + e + 8 * hi;
        int s = s1 + m;
        float val = (s <= t) ? __expf(cat[e] - cs1) * g1[e] : 0.f;
        Mtile[(e + 8 * hi) * 32 + 16 + m] = to_bf16(val);
      }
    } else {
#pragma unroll
      for (int e = 0; e < 8; ++e)
        Mtile[(e + 8 * hi) * 32 + 16 + m] = to_bf16(0.f);
    }
    // same-wave LDS is in-order; waits + memory clobber stop compiler reordering
    asm volatile("s_wait_dscnt 0" ::: "memory");        // Mtile stores done
    asm volatile("s_wait_tensorcnt 0x0" ::: "memory");  // Xtile DMA done

    v16bf mf = load_frag(Mtile, 32, 0, 0);         // A-fragment of M from LDS
#pragma unroll
    for (int j = 0; j < 8; ++j) {
      v16bf xb = load_frag(Xtile, 32, j * 16, 0);  // B-fragments from LDS tile
      acc[j] = wmma_bf16(mf, xb, acc[j]);
    }
  }

  float Dh = Dvec[h];
#pragma unroll
  for (int j = 0; j < 8; ++j)
#pragma unroll
    for (int e = 0; e < 8; ++e) {
      int t = tbase + e + 8 * hi;
      int c = h * D_HEAD + j * 16 + m;
      y[(size_t)t * D_INNER + c] = acc[j][e] + Dh * xBCc[(size_t)t * CONV_DIM + c];
    }
}

__global__ void k_gate_norm(const float* y, const float* zx, const float* w,
                            __bf16* out) {
  int t = blockIdx.x, tid = threadIdx.x;
  float v[8]; float ss = 0.f;
#pragma unroll
  for (int i = 0; i < 8; ++i) {
    int c = tid + i * 256;
    float z = zx[(size_t)t * D_WIN + c];
    float g = z / (1.f + __expf(-z));
    float val = y[(size_t)t * D_INNER + c] * g;
    v[i] = val; ss += val * val;
  }
  __shared__ float red[256];
  red[tid] = ss; __syncthreads();
  for (int off = 128; off > 0; off >>= 1) {
    if (tid < off) red[tid] += red[tid + off];
    __syncthreads();
  }
  float inv = rsqrtf(red[0] / (float)D_INNER + 1e-5f);
#pragma unroll
  for (int i = 0; i < 8; ++i) {
    int c = tid + i * 256;
    out[(size_t)t * D_INNER + c] = to_bf16(v[i] * inv * w[c]);
  }
}

// ---------- launcher ---------------------------------------------------------

extern "C" void kernel_launch(void* const* d_in, const int* in_sizes, int n_in,
                              void* d_out, int out_size, void* d_ws, size_t ws_size,
                              hipStream_t stream) {
  const float* x        = (const float*)d_in[0];
  const float* norm_in  = (const float*)d_in[1];
  const float* norm_out = (const float*)d_in[2];
  const float* win_w    = (const float*)d_in[3];
  const float* wout_w   = (const float*)d_in[4];
  const float* conv     = (const float*)d_in[5];
  const float* conv_b   = (const float*)d_in[6];
  const float* dt_bias  = (const float*)d_in[7];
  const float* A_log    = (const float*)d_in[8];
  const float* Dvec     = (const float*)d_in[9];
  float* out = (float*)d_out;

  char* wp = (char*)d_ws;
  auto alloc = [&](size_t bytes) {
    void* p = (void*)wp;
    wp += (bytes + 255) & ~(size_t)255;
    return p;
  };
  __bf16* xn_bf   = (__bf16*)alloc((size_t)L_SEQ * D_MODEL * 2);
  __bf16* win_bf  = (__bf16*)alloc((size_t)D_WIN * D_MODEL * 2);
  __bf16* wout_bf = (__bf16*)alloc((size_t)D_MODEL * D_INNER * 2);
  float*  zx      = (float*) alloc((size_t)L_SEQ * D_WIN * 4);
  float*  xBCc    = (float*) alloc((size_t)L_SEQ * CONV_DIM * 4);
  float*  dt_sp   = (float*) alloc((size_t)L_SEQ * N_HEADS * 4);
  float*  cumA    = (float*) alloc((size_t)N_HEADS * L_SEQ * 4);
  __bf16* Bb      = (__bf16*)alloc((size_t)L_SEQ * D_STATE * 2);
  __bf16* Cb      = (__bf16*)alloc((size_t)L_SEQ * D_STATE * 2);
  __bf16* xdtT    = (__bf16*)alloc((size_t)D_INNER * L_SEQ * 2);
  float*  ybuf    = (float*) alloc((size_t)L_SEQ * D_INNER * 4);
  __bf16* ygn     = (__bf16*)alloc((size_t)L_SEQ * D_INNER * 2);

  k_rmsnorm_in<<<L_SEQ, 256, 0, stream>>>(x, norm_in, xn_bf);

  int nwin = D_WIN * D_MODEL;
  k_f32_to_bf16<<<(nwin + 255) / 256, 256, 0, stream>>>(win_w, win_bf, nwin);
  int nwout = D_MODEL * D_INNER;
  k_f32_to_bf16<<<(nwout + 255) / 256, 256, 0, stream>>>(wout_w, wout_bf, nwout);

  // in-proj: 2048x4240 = xn(2048x1024) @ win^T ; 265 N-tiles -> 67 blocks of 4
  k_gemm_bf16<<<dim3(32, 67), 128, 0, stream>>>(xn_bf, win_bf, zx,
                                                L_SEQ, D_WIN, D_MODEL);

  int nconv = L_SEQ * CONV_DIM;
  k_conv_silu<<<(nconv + 255) / 256, 256, 0, stream>>>(zx, conv, conv_b, xBCc);
  k_dt_cum<<<1, 16, 0, stream>>>(zx, dt_bias, A_log, dt_sp, cumA);
  k_prep<<<(nconv + 255) / 256, 256, 0, stream>>>(xBCc, dt_sp, xdtT, Bb, Cb);

  k_ssm<<<dim3(L_SEQ / 16, N_HEADS), 32, 0, stream>>>(Cb, Bb, xdtT, cumA, Dvec,
                                                      xBCc, ybuf);

  k_gate_norm<<<L_SEQ, 256, 0, stream>>>(ybuf, zx, norm_out, ygn);

  // out-proj: 2048x1024 = ygn(2048x2048) @ wout^T ; 64 N-tiles -> 16 blocks
  k_gemm_bf16<<<dim3(32, 16), 128, 0, stream>>>(ygn, wout_bf, out,
                                                L_SEQ, D_MODEL, D_INNER);

  (void)in_sizes; (void)n_in; (void)out_size; (void)ws_size;
}
